// Bidirectional_74448963109339
// MI455X (gfx1250) — compile-verified, matrix operand order checked by reading, not run on hardware
//
#include <hip/hip_runtime.h>
#include <hip/hip_bf16.h>
#include <math.h>

typedef __attribute__((ext_vector_type(16))) __bf16 v16bf;
typedef __attribute__((ext_vector_type(8)))  float  v8f;
typedef __attribute__((ext_vector_type(4)))  unsigned int u32x4;
typedef __attribute__((ext_vector_type(8)))  int i32x8;
typedef __attribute__((ext_vector_type(4)))  int i32x4;

#define Bn 64
#define Tn 512
#define Fn 128
#define Hn 512
#define G3 1536
#define On 8
#define BN_EPS 1e-3f

__device__ __forceinline__ unsigned short f2bf(float f) {
  union { __bf16 h; unsigned short s; } u; u.h = (__bf16)f; return u.s;
}
__device__ __forceinline__ float bf2f(unsigned short s) {
  union { __bf16 h; unsigned short s; } u; u.s = s; return (float)u.h;
}

// Repack a row-major fp32 [K,N] weight into bf16 WMMA-B tile order:
// tile (nt, kt) = 32 lanes x 16 contiguous bf16 per lane.
__global__ void pack_b(const float* __restrict__ src, unsigned short* __restrict__ dst,
                       int K, int N) {
  int idx = blockIdx.x * blockDim.x + threadIdx.x;
  if (idx >= K * N) return;
  int k = idx / N, n = idx - k * N;
  int nt = n >> 4, nl = n & 15;
  int kt = k >> 5, kin = k & 31;
  int lane = ((kin >> 4) << 4) | nl;
  int e = kin & 15;
  int KT = K >> 5;
  size_t d = (((size_t)nt * KT + kt) * 32 + lane) * 16 + e;
  dst[d] = f2bf(src[idx]);
}

// h = BN1(x @ w_in + b_in)  -> bf16 [B*T, H]
__global__ __launch_bounds__(256) void k_gemm_in(
    const float* __restrict__ x, const unsigned short* __restrict__ wp,
    const float* __restrict__ b_in, const float* __restrict__ g1,
    const float* __restrict__ be1, const float* __restrict__ m1,
    const float* __restrict__ v1, unsigned short* __restrict__ hbf) {
  const int lane  = threadIdx.x & 31;
  const int wv    = threadIdx.x >> 5;           // 0..7
  const int mtile = blockIdx.x;                 // 0..2047
  const int ntile = blockIdx.y * 8 + wv;        // 0..31
  const int row   = mtile * 16 + (lane & 15);
  const int KT    = Fn / 32;                    // 4
  v8f c = {};
#pragma unroll
  for (int kt = 0; kt < KT; ++kt) {
    const float* ap = x + (size_t)row * Fn + kt * 32 + ((lane >> 4) << 4);
    union { v16bf v; __bf16 h[16]; } a;
#pragma unroll
    for (int i = 0; i < 16; ++i) a.h[i] = (__bf16)ap[i];
    v16bf b = *(const v16bf*)(wp + ((((size_t)ntile * KT + kt) * 32 + lane) << 4));
    c = __builtin_amdgcn_wmma_f32_16x16x32_bf16(false, a.v, false, b, (short)0, c,
                                                false, false);
  }
  const int col = ntile * 16 + (lane & 15);
  float s1 = g1[col] * rsqrtf(v1[col] + BN_EPS);
  float t1 = (b_in[col] - m1[col]) * s1 + be1[col];
#pragma unroll
  for (int i = 0; i < 8; ++i) {
    int r = mtile * 16 + i + ((lane >> 4) << 3);
    hbf[(size_t)r * Hn + col] = f2bf(c[i] * s1 + t1);
  }
}

// xp_dir = h @ wx_dir + b_dir[0]  -> bf16 [B*T, 3H], both directions via grid.z
__global__ __launch_bounds__(256) void k_gemm_xp(
    const unsigned short* __restrict__ hbf,
    const unsigned short* __restrict__ wxf_p, const unsigned short* __restrict__ wxb_p,
    const float* __restrict__ bfv, const float* __restrict__ bbv,
    unsigned short* __restrict__ xpf, unsigned short* __restrict__ xpb) {
  const int lane  = threadIdx.x & 31;
  const int wv    = threadIdx.x >> 5;
  const int mtile = blockIdx.x;                 // 0..2047
  const int ntile = blockIdx.y * 8 + wv;        // 0..95
  const int dir   = blockIdx.z;
  const unsigned short* wp = dir ? wxb_p : wxf_p;
  const float* bias        = dir ? bbv : bfv;   // row 0: input bias
  unsigned short* xp       = dir ? xpb : xpf;
  const int row = mtile * 16 + (lane & 15);
  const int KT  = Hn / 32;                      // 16
  v8f c = {};
#pragma unroll 4
  for (int kt = 0; kt < KT; ++kt) {
    v16bf a = *(const v16bf*)(hbf + (size_t)row * Hn + kt * 32 + ((lane >> 4) << 4));
    v16bf b = *(const v16bf*)(wp + ((((size_t)ntile * KT + kt) * 32 + lane) << 4));
    c = __builtin_amdgcn_wmma_f32_16x16x32_bf16(false, a, false, b, (short)0, c,
                                                false, false);
  }
  const int col = ntile * 16 + (lane & 15);
  float bv = bias[col];
#pragma unroll
  for (int i = 0; i < 8; ++i) {
    int r = mtile * 16 + i + ((lane >> 4) << 3);
    xp[(size_t)r * G3 + col] = f2bf(c[i] + bv);
  }
}

// TDM: DMA a 16-row x 1536-col bf16 tile (row stride Tn*G3 elements) into LDS.
// D# group0: count=1, lds_addr, 57-bit global addr, type=2.
// D# group1: data_size=2B, tensor_dim0=1536, tensor_dim1=16, tile=1536x16,
//            tensor_dim0_stride = Tn*G3.
__device__ __forceinline__ void tdm_load_tile(const unsigned short* gptr,
                                              unsigned lds_off) {
  unsigned long long ga = (unsigned long long)(uintptr_t)gptr;
  u32x4 g0;
  g0.x = 1u;                                   // count=1, user desc
  g0.y = lds_off;                              // LDS byte address
  g0.z = (unsigned)ga;                         // global addr [31:0]
  g0.w = (unsigned)(ga >> 32) | (2u << 30);    // addr [56:32] | type=2
  i32x8 g1;
  g1[0] = 0x00010000;                          // data_size=1 (2 bytes)
  g1[1] = 0x06000000;                          // tensor_dim0=1536 (bits 48..63)
  g1[2] = 0x00100000;                          // tensor_dim1=16   (bits 80..95)
  g1[3] = 0x06000000;                          // tile_dim0=1536   (bits 112..127)
  g1[4] = 0x00000010;                          // tile_dim1=16, tile_dim2=0
  g1[5] = (int)(Tn * G3);                      // dim0_stride lo32 = 786432
  g1[6] = 0;
  g1[7] = 0;
  i32x4 z4 = {0, 0, 0, 0};
  i32x8 z8 = {0, 0, 0, 0, 0, 0, 0, 0};
  // 6-arg variant (clang-23 / therock-10.0 headers): groups 2,3 unused (2D tile)
  __builtin_amdgcn_tensor_load_to_lds(g0, g1, z4, z4, z8, 0);
}

// Persistent GRU scan: 8 blocks = {fwd,bwd} x 4 batch slices of 16 rows.
// Per step: 16 waves x 6 N-tiles x 16 K-steps of v_wmma_f32_16x16x32_bf16,
// with TDM double-buffered prefetch of the next step's xp tile into LDS.
__global__ __launch_bounds__(512) void k_scan(
    const unsigned short* __restrict__ xpf, const unsigned short* __restrict__ xpb,
    const unsigned short* __restrict__ whf_p, const unsigned short* __restrict__ whb_p,
    const float* __restrict__ bfv, const float* __restrict__ bbv,
    unsigned short* __restrict__ outf, unsigned short* __restrict__ outb) {
  __shared__ float          hsm[16 * Hn];        // fp32 state, 32 KB
  __shared__ unsigned short hbl[16 * Hn];        // bf16 state (WMMA A feed), 16 KB
  __shared__ float          rec[16 * G3];        // recurrent GEMM out, 96 KB
  __shared__ unsigned short xpl[2][16 * G3];     // TDM xp double buffer, 96 KB
  const int tid  = threadIdx.x;
  const int lane = tid & 31;
  const int wv   = tid >> 5;                     // 0..15
  const int dir  = blockIdx.x >> 2;
  const int b0   = (blockIdx.x & 3) * 16;
  const unsigned short* xp = dir ? xpb : xpf;
  const unsigned short* wp = dir ? whb_p : whf_p;
  const float* brec        = (dir ? bbv : bfv) + G3;   // row 1: recurrent bias
  unsigned short* out      = dir ? outb : outf;

  // per-wave recurrent-bias registers: column nt*16 + (lane&15), fixed all steps
  float cbias[6];
#pragma unroll
  for (int j = 0; j < 6; ++j) cbias[j] = brec[(wv * 6 + j) * 16 + (lane & 15)];

  for (int i = tid; i < 16 * Hn; i += 512) { hsm[i] = 0.0f; hbl[i] = 0; }

  // prefetch step 0's xp tile
  if (wv == 0) {
    int t0 = dir ? (Tn - 1) : 0;
    tdm_load_tile(xp + (size_t)(b0 * Tn + t0) * G3,
                  (unsigned)(uintptr_t)&xpl[0][0]);
  }
  __syncthreads();

  const int KT = Hn / 32;                        // 16
  for (int s = 0; s < Tn; ++s) {
    const int t = dir ? (Tn - 1 - s) : s;
    // kick off next step's xp tile DMA while the recurrent GEMM runs
    if (wv == 0 && s + 1 < Tn) {
      int tnx = dir ? (Tn - 2 - s) : (s + 1);
      tdm_load_tile(xp + (size_t)(b0 * Tn + tnx) * G3,
                    (unsigned)(uintptr_t)&xpl[(s + 1) & 1][0]);
    }
    // rec = h_state @ wh + b_rec  (bf16 WMMA, fp32 accumulate)
#pragma unroll
    for (int j = 0; j < 6; ++j) {
      const int nt = wv * 6 + j;                 // 0..95
      v8f c = {};
#pragma unroll 4
      for (int kt = 0; kt < KT; ++kt) {
        v16bf a = *(const v16bf*)(&hbl[(lane & 15) * Hn + kt * 32 + ((lane >> 4) << 4)]);
        v16bf b = *(const v16bf*)(wp + ((((size_t)nt * KT + kt) * 32 + lane) << 4));
        c = __builtin_amdgcn_wmma_f32_16x16x32_bf16(false, a, false, b, (short)0, c,
                                                    false, false);
      }
#pragma unroll
      for (int i = 0; i < 8; ++i)
        rec[(i + ((lane >> 4) << 3)) * G3 + nt * 16 + (lane & 15)] = c[i] + cbias[j];
    }
    __syncthreads();
    // TDM ops are per-wave in-order: <=1 outstanding means current tile landed
    if (wv == 0) {
      if (s + 1 < Tn) __builtin_amdgcn_s_wait_tensorcnt(1);
      else            __builtin_amdgcn_s_wait_tensorcnt(0);
    }
    __syncthreads();
    // fused gate update (keras reset_after=True, gate order z,r,h)
    {
      const unsigned short* xt = &xpl[s & 1][0];
      const int rr = tid >> 5;                   // 0..15 batch row in slice
      const int c0 = (tid & 31) * 16;
      const size_t orow = ((size_t)(b0 + rr) * Tn + t) * Hn;
#pragma unroll 4
      for (int j = 0; j < 16; ++j) {
        const int n = c0 + j;
        float xz = bf2f(xt[rr * G3 + n]);
        float xr = bf2f(xt[rr * G3 + Hn + n]);
        float xh = bf2f(xt[rr * G3 + 2 * Hn + n]);
        float rz = rec[rr * G3 + n];
        float rg = rec[rr * G3 + Hn + n];
        float rh = rec[rr * G3 + 2 * Hn + n];
        float z  = 1.0f / (1.0f + __expf(-(xz + rz)));
        float r  = 1.0f / (1.0f + __expf(-(xr + rg)));
        float hh = tanhf(xh + r * rh);
        float hn = z * hsm[rr * Hn + n] + (1.0f - z) * hh;
        hsm[rr * Hn + n] = hn;
        unsigned short hb = f2bf(hn);
        hbl[rr * Hn + n] = hb;
        out[orow + n] = hb;
      }
    }
    __syncthreads();
  }
}

// y = BN2(concat(fwd,bwd) @ w_out + b_out), O=8 (memory-bound; w_out in LDS)
__global__ __launch_bounds__(256) void k_out(
    const unsigned short* __restrict__ outf, const unsigned short* __restrict__ outb,
    const float* __restrict__ w_out, const float* __restrict__ b_out,
    const float* __restrict__ g2, const float* __restrict__ be2,
    const float* __restrict__ m2, const float* __restrict__ v2,
    float* __restrict__ y) {
  __shared__ float wsh[2 * Hn * On];             // 32 KB
  for (int i = threadIdx.x; i < 2 * Hn * On; i += 256) wsh[i] = w_out[i];
  __syncthreads();
  const int m = blockIdx.x * 256 + threadIdx.x;  // 0..32767
  float acc[On] = {};
  const unsigned short* pf = outf + (size_t)m * Hn;
  const unsigned short* pb = outb + (size_t)m * Hn;
  for (int k = 0; k < Hn; ++k) {
    float a = bf2f(pf[k]);
#pragma unroll
    for (int o = 0; o < On; ++o) acc[o] += a * wsh[k * On + o];
  }
  for (int k = 0; k < Hn; ++k) {
    float a = bf2f(pb[k]);
#pragma unroll
    for (int o = 0; o < On; ++o) acc[o] += a * wsh[(Hn + k) * On + o];
  }
#pragma unroll
  for (int o = 0; o < On; ++o) {
    float sc = g2[o] * rsqrtf(v2[o] + BN_EPS);
    y[(size_t)m * On + o] = (acc[o] + b_out[o] - m2[o]) * sc + be2[o];
  }
}

extern "C" void kernel_launch(void* const* d_in, const int* in_sizes, int n_in,
                              void* d_out, int out_size, void* d_ws, size_t ws_size,
                              hipStream_t stream) {
  const float* x     = (const float*)d_in[0];
  const float* w_in  = (const float*)d_in[1];
  const float* b_in  = (const float*)d_in[2];
  const float* g1    = (const float*)d_in[3];
  const float* be1   = (const float*)d_in[4];
  const float* m1    = (const float*)d_in[5];
  const float* v1    = (const float*)d_in[6];
  const float* wxf   = (const float*)d_in[7];
  const float* whf   = (const float*)d_in[8];
  const float* bfv   = (const float*)d_in[9];
  const float* wxb   = (const float*)d_in[10];
  const float* whb   = (const float*)d_in[11];
  const float* bbv   = (const float*)d_in[12];
  const float* w_out = (const float*)d_in[13];
  const float* b_out = (const float*)d_in[14];
  const float* g2    = (const float*)d_in[15];
  const float* be2   = (const float*)d_in[16];
  const float* m2    = (const float*)d_in[17];
  const float* v2    = (const float*)d_in[18];
  float* y = (float*)d_out;

  char* ws = (char*)d_ws;
  size_t off = 0;
  auto take = [&](size_t bytes) {
    char* p = ws + off;
    off = (off + bytes + 255) & ~(size_t)255;
    return p;
  };
  unsigned short* w_in_p = (unsigned short*)take((size_t)Fn * Hn * 2);
  unsigned short* wxf_p  = (unsigned short*)take((size_t)Hn * G3 * 2);
  unsigned short* whf_p  = (unsigned short*)take((size_t)Hn * G3 * 2);
  unsigned short* wxb_p  = (unsigned short*)take((size_t)Hn * G3 * 2);
  unsigned short* whb_p  = (unsigned short*)take((size_t)Hn * G3 * 2);
  unsigned short* hbf    = (unsigned short*)take((size_t)Bn * Tn * Hn * 2);
  unsigned short* xpf    = (unsigned short*)take((size_t)Bn * Tn * G3 * 2);
  unsigned short* xpb    = (unsigned short*)take((size_t)Bn * Tn * G3 * 2);
  unsigned short* outf   = (unsigned short*)take((size_t)Bn * Tn * Hn * 2);
  unsigned short* outb   = (unsigned short*)take((size_t)Bn * Tn * Hn * 2);

  pack_b<<<(Fn * Hn + 255) / 256, 256, 0, stream>>>(w_in, w_in_p, Fn, Hn);
  pack_b<<<(Hn * G3 + 255) / 256, 256, 0, stream>>>(wxf, wxf_p, Hn, G3);
  pack_b<<<(Hn * G3 + 255) / 256, 256, 0, stream>>>(whf, whf_p, Hn, G3);
  pack_b<<<(Hn * G3 + 255) / 256, 256, 0, stream>>>(wxb, wxb_p, Hn, G3);
  pack_b<<<(Hn * G3 + 255) / 256, 256, 0, stream>>>(whb, whb_p, Hn, G3);

  k_gemm_in<<<dim3(2048, 4), 256, 0, stream>>>(x, w_in_p, b_in, g1, be1, m1, v1, hbf);
  k_gemm_xp<<<dim3(2048, 12, 2), 256, 0, stream>>>(hbf, wxf_p, wxb_p, bfv, bbv, xpf, xpb);
  k_scan<<<8, 512, 0, stream>>>(xpf, xpb, whf_p, whb_p, bfv, bbv, outf, outb);
  k_out<<<128, 256, 0, stream>>>(outf, outb, w_out, b_out, g2, be2, m2, v2, y);
}
